// MPNN_ENN_K_Sum_5076651344423
// MI455X (gfx1250) — matrix operand order repaired and checked by reading, not compiled, not run on hardware
//
#include <hip/hip_runtime.h>
#include <hip/hip_bf16.h>

#define N_NODES  10000
#define N_EDGES  64000
#define N_GRAPHS 64
#define HDIM     64
#define TSTEPS   3

typedef __attribute__((ext_vector_type(16))) __bf16 v16bf;
typedef __attribute__((ext_vector_type(8)))  __bf16 bhalf8;
typedef __attribute__((ext_vector_type(8)))  float  v8f;

#define ATILE_STRIDE 68   // 16-row A tile in LDS, padded: 68*4B = 272B row stride (4-bank rotate)

// ---------------------------------------------------------------------------
// WMMA fragment loaders (layouts per CDNA5 ISA 7.12.2, wave32)
// A: 16x32 bf16, lane = {half = lane>>4, row = lane&15};
//    element j of v16bf: k = kbase + (j<8 ? half*8 + j : 16 + half*8 + (j-8))
//    -> two contiguous 16B chunks per lane, converted fp32->bf16 on load.
static __device__ __forceinline__ v16bf load_a_frag_f32(const float* __restrict__ A,
                                                        int lda, int m0, int kbase, int lane) {
  const int row  = lane & 15;
  const int half = lane >> 4;
  const float* p = A + (size_t)(m0 + row) * lda + kbase + half * 8;
  v16bf a;
#pragma unroll
  for (int j = 0; j < 8; ++j) a[j] = (__bf16)p[j];
#pragma unroll
  for (int j = 0; j < 8; ++j) a[8 + j] = (__bf16)p[16 + j];
  return a;
}

// Same fragment, sourced from the LDS-staged A tile (padded row stride).
static __device__ __forceinline__ v16bf load_a_frag_lds(const float* at, int kbase, int lane) {
  const int row  = lane & 15;
  const int half = lane >> 4;
  const float* p = at + row * ATILE_STRIDE + kbase + half * 8;
  v16bf a;
#pragma unroll
  for (int j = 0; j < 8; ++j) a[j] = (__bf16)p[j];
#pragma unroll
  for (int j = 0; j < 8; ++j) a[8 + j] = (__bf16)p[16 + j];
  return a;
}

// ---------------------------------------------------------------------------
// Swizzle W[64, N] fp32 -> bf16 fragment layout: Bsw[ntile][kstep][lane][j]
// with k = kstep*32 + (j<8 ? (lane>>4)*8 + j : 16 + (lane>>4)*8 + (j-8)), col = ntile*16 + (lane&15)
__global__ void swizzle_b_kernel(const float* __restrict__ W, __bf16* __restrict__ Bsw,
                                 int N, int Ntiles) {
  int idx = blockIdx.x * blockDim.x + threadIdx.x;
  int total = Ntiles * 1024;                      // Ntiles * 2 * 32 * 16
  if (idx >= total) return;
  int j     = idx & 15;
  int lane  = (idx >> 4) & 31;
  int kstep = (idx >> 9) & 1;
  int ntile = idx >> 10;
  int k   = kstep * 32 + ((lane >> 4) << 3) + (j & 7) + ((j >> 3) << 4);
  int col = (ntile << 4) + (lane & 15);
  Bsw[idx] = (__bf16)W[(size_t)k * N + col];
}

// ---------------------------------------------------------------------------
// Generic K=64 WMMA GEMM: C[M,N] = A[M,64](fp32, cvt->bf16) @ Bsw(bf16) + bias
// One wave per 16x(16*NT) tile; fp32 accumulation; output fp32 or bf16.
// Requires (N/16) % NT == 0; groups = (N/16)/NT.
// STAGE_A: all 8 waves of a 256-thread block share one mtile (requires
// groups % 8 == 0 and an exactly-divisible grid); the 4KB A tile is staged
// into LDS once per block with GLOBAL_LOAD_ASYNC_TO_LDS_B128 (ASYNCcnt path).
// Per k-step: all NT B fragments are loaded first (one base address, +2048B
// immediate offsets -> clustered b128 loads), then NT back-to-back WMMAs.
template <int NT, bool OUT_BF16, bool STAGE_A>
__global__ void gemm_k64_wmma(const float* __restrict__ A, const __bf16* __restrict__ Bsw,
                              const float* __restrict__ bias, void* __restrict__ Cout,
                              int M, int N, int groups) {
  const int lane = threadIdx.x & 31;
  __shared__ float atile[16 * ATILE_STRIDE];

  if (STAGE_A) {
    // One b128 async lane-transfer per thread: 256 * 16B = 4KB = full 16x64 fp32 tile.
    const int tid   = threadIdx.x;
    const int mtblk = __builtin_amdgcn_readfirstlane((int)((blockIdx.x * 8) / groups));
    const int row   = tid >> 4;       // 0..15
    const int chunk = tid & 15;       // 16B chunks within a 256B row
    const char* gsrc = (const char*)(A + (size_t)(mtblk * 16 + row) * 64) + chunk * 16;
    // Low 32 bits of a generic pointer to __shared__ are the LDS byte offset.
    unsigned ldst = (unsigned)(size_t)(&atile[row * ATILE_STRIDE]) + (unsigned)(chunk * 16);
    asm volatile("global_load_async_to_lds_b128 %0, %1, off"
                 :: "v"(ldst), "v"(gsrc) : "memory");
    asm volatile("s_wait_asynccnt 0x0" ::: "memory");
    __syncthreads();
  }

  const int wid  = __builtin_amdgcn_readfirstlane(
      (int)((blockIdx.x * blockDim.x + threadIdx.x) >> 5));
  const int Mtiles = M >> 4;
  if (wid >= Mtiles * groups) return;          // scalar branch; full-wave exit
  const int mtile = wid / groups;
  const int ng    = wid % groups;
  const int m0    = mtile << 4;
  const int nbase = ng * NT;

  v8f zero = {0.f, 0.f, 0.f, 0.f, 0.f, 0.f, 0.f, 0.f};
  v8f acc[NT];
#pragma unroll
  for (int t = 0; t < NT; ++t) acc[t] = zero;

#pragma unroll
  for (int ks = 0; ks < 2; ++ks) {
    v16bf a = STAGE_A ? load_a_frag_lds(atile, ks * 32, lane)
                      : load_a_frag_f32(A, 64, m0, ks * 32, lane);
    // One per-lane base address; consecutive n-tiles are +1024 elements (2048B),
    // expressible as instruction immediate offsets -> clause of b128 loads.
    const __bf16* bbase = Bsw + (((size_t)(nbase * 2 + ks) * 32 + lane) << 4);
    v16bf b[NT];
#pragma unroll
    for (int t = 0; t < NT; ++t) b[t] = *(const v16bf*)(bbase + (t << 10));
#pragma unroll
    for (int t = 0; t < NT; ++t)
      acc[t] = __builtin_amdgcn_wmma_f32_16x16x32_bf16(
          false, a, false, b[t], (short)0, acc[t], false, false);
  }

  const int col  = lane & 15;
  const int half = lane >> 4;
#pragma unroll
  for (int t = 0; t < NT; ++t) {
    int n = ((nbase + t) << 4) + col;
    float bv = bias[n];
    if (OUT_BF16) {
      __bf16* C = (__bf16*)Cout;
#pragma unroll
      for (int r = 0; r < 8; ++r)
        C[(size_t)(m0 + r + 8 * half) * N + n] = (__bf16)(acc[t][r] + bv);
    } else {
      float* C = (float*)Cout;
#pragma unroll
      for (int r = 0; r < 8; ++r)
        C[(size_t)(m0 + r + 8 * half) * N + n] = acc[t][r] + bv;
    }
  }
}

// ---------------------------------------------------------------------------
__global__ void fill_f32(float* __restrict__ p, float v, int n) {
  int i = blockIdx.x * blockDim.x + threadIdx.x;
  if (i < n) p[i] = v;
}

// h = node_features @ W_in + b_in   (K=32, tiny)
__global__ void input_proj(const float* __restrict__ X, const float* __restrict__ W,
                           const float* __restrict__ b, float* __restrict__ h) {
  int idx = blockIdx.x * blockDim.x + threadIdx.x;
  if (idx >= N_NODES * HDIM) return;
  int n = idx >> 6, j = idx & 63;
  float acc = b[j];
#pragma unroll
  for (int k = 0; k < 32; ++k) acc = fmaf(X[n * 32 + k], W[k * 64 + j], acc);
  h[idx] = acc;
}

// e = relu(edge_features @ W_e1 + b_e1)   (K=16, tiny)
__global__ void edge_enc1(const float* __restrict__ EF, const float* __restrict__ W,
                          const float* __restrict__ b, float* __restrict__ e) {
  int idx = blockIdx.x * blockDim.x + threadIdx.x;
  if (idx >= N_EDGES * HDIM) return;
  int n = idx >> 6, j = idx & 63;
  float acc = b[j];
#pragma unroll
  for (int k = 0; k < 16; ++k) acc = fmaf(EF[n * 16 + k], W[k * 64 + j], acc);
  e[idx] = fmaxf(acc, 0.f);
}

// One wave per edge: msg = edge_mat[e] (64x64 bf16) @ h[Esrc[e]], scatter-add to m[Etgt[e]]
__global__ void msg_kernel(const __bf16* __restrict__ emat, const float* __restrict__ h,
                           const int* __restrict__ Esrc, const int* __restrict__ Etgt,
                           float* __restrict__ m) {
  const int lane = threadIdx.x & 31;
  const int e    = __builtin_amdgcn_readfirstlane(
      (int)((blockIdx.x * blockDim.x + threadIdx.x) >> 5));
  if (e >= N_EDGES) return;
  const int src = __builtin_amdgcn_readfirstlane(Esrc[e]);   // scalar gather index
  const float hs0 = h[(size_t)src * 64 + lane];        // k = lane
  const float hs1 = h[(size_t)src * 64 + 32 + lane];   // k = lane + 32
  const __bf16* row0 = emat + (size_t)e * 4096 + (size_t)lane * 64;        // out row lane
  const __bf16* row1 = emat + (size_t)e * 4096 + (size_t)(lane + 32) * 64; // out row lane+32
  __builtin_prefetch(row0, 0, 1);
  __builtin_prefetch(row1, 0, 1);
  float acc0 = 0.f, acc1 = 0.f;
#pragma unroll
  for (int c = 0; c < 8; ++c) {
    bhalf8 c0 = *(const bhalf8*)(row0 + c * 8);
    bhalf8 c1 = *(const bhalf8*)(row1 + c * 8);
#pragma unroll
    for (int j = 0; j < 8; ++j) {
      int k = c * 8 + j;
      float hv = (k < 32) ? __shfl(hs0, k, 32) : __shfl(hs1, k - 32, 32);
      acc0 = fmaf((float)c0[j], hv, acc0);
      acc1 = fmaf((float)c1[j], hv, acc1);
    }
  }
  const int tgt = __builtin_amdgcn_readfirstlane(Etgt[e]);
  atomicAdd(&m[(size_t)tgt * 64 + lane], acc0);
  atomicAdd(&m[(size_t)tgt * 64 + 32 + lane], acc1);
}

// GRU elementwise update from precomputed gi = m@W_ih+b_ih, gh = h@W_hh+b_hh
__global__ void gru_update(const float* __restrict__ gi, const float* __restrict__ gh,
                           float* __restrict__ h) {
  int idx = blockIdx.x * blockDim.x + threadIdx.x;
  if (idx >= N_NODES * HDIM) return;
  int n = idx >> 6, j = idx & 63;
  const float* gin = gi + (size_t)n * 192;
  const float* ghn = gh + (size_t)n * 192;
  float r  = 1.f / (1.f + __expf(-(gin[j]       + ghn[j])));
  float z  = 1.f / (1.f + __expf(-(gin[64 + j]  + ghn[64 + j])));
  float nn = tanhf(gin[128 + j] + r * ghn[128 + j]);
  float hv = h[idx];
  h[idx] = (1.f - z) * nn + z * hv;
}

// y = h @ W_out + b_out; out[batch[n]] += y
__global__ void readout(const float* __restrict__ h, const float* __restrict__ Wout,
                        const float* __restrict__ bout, const int* __restrict__ batch,
                        float* __restrict__ out) {
  int n = blockIdx.x * blockDim.x + threadIdx.x;
  if (n >= N_NODES) return;
  float acc = bout[0];
#pragma unroll
  for (int k = 0; k < 64; ++k) acc = fmaf(h[(size_t)n * 64 + k], Wout[k], acc);
  atomicAdd(&out[batch[n]], acc);
}

// ---------------------------------------------------------------------------
extern "C" void kernel_launch(void* const* d_in, const int* in_sizes, int n_in,
                              void* d_out, int out_size, void* d_ws, size_t ws_size,
                              hipStream_t stream) {
  const float* node_f = (const float*)d_in[0];
  const float* edge_f = (const float*)d_in[1];
  const int*   Esrc   = (const int*)d_in[2];
  const int*   Etgt   = (const int*)d_in[3];
  const int*   batch  = (const int*)d_in[4];
  const float* W_in   = (const float*)d_in[5];
  const float* b_in   = (const float*)d_in[6];
  const float* W_e1   = (const float*)d_in[7];
  const float* b_e1   = (const float*)d_in[8];
  const float* W_e2   = (const float*)d_in[9];
  const float* b_e2   = (const float*)d_in[10];
  const float* W_ih   = (const float*)d_in[11];
  const float* W_hh   = (const float*)d_in[12];
  const float* b_ih   = (const float*)d_in[13];
  const float* b_hh   = (const float*)d_in[14];
  const float* W_out  = (const float*)d_in[15];
  const float* b_out  = (const float*)d_in[16];
  float* out = (float*)d_out;
  (void)in_sizes; (void)n_in; (void)out_size; (void)ws_size;

  // workspace carve-up (256B aligned slices)
  size_t off = 0;
  auto take = [&](size_t bytes) {
    void* p = (char*)d_ws + off;
    off += (bytes + 255) & ~(size_t)255;
    return p;
  };
  float*  e_act = (float*) take((size_t)N_EDGES * 64 * 4);      // 16.4 MB
  __bf16* We2sw = (__bf16*)take((size_t)256 * 1024 * 2);        // 512 KB
  __bf16* Wihsw = (__bf16*)take((size_t)12 * 1024 * 2);
  __bf16* Whhsw = (__bf16*)take((size_t)12 * 1024 * 2);
  float*  hbuf  = (float*) take((size_t)N_NODES * 64 * 4);
  float*  mbuf  = (float*) take((size_t)N_NODES * 64 * 4);
  float*  gi    = (float*) take((size_t)N_NODES * 192 * 4);
  float*  gh    = (float*) take((size_t)N_NODES * 192 * 4);
  __bf16* emat  = (__bf16*)take((size_t)N_EDGES * 4096 * 2);    // 524 MB

  const int B = 256;

  // weight swizzles (bf16 fragment layout)
  swizzle_b_kernel<<<(256 * 1024 + B - 1) / B, B, 0, stream>>>(W_e2, We2sw, 4096, 256);
  swizzle_b_kernel<<<(12 * 1024 + B - 1) / B, B, 0, stream>>>(W_ih, Wihsw, 192, 12);
  swizzle_b_kernel<<<(12 * 1024 + B - 1) / B, B, 0, stream>>>(W_hh, Whhsw, 192, 12);

  // h0 = node_features @ W_in + b_in
  input_proj<<<(N_NODES * 64 + B - 1) / B, B, 0, stream>>>(node_f, W_in, b_in, hbuf);

  // e = relu(edge_features @ W_e1 + b_e1)
  edge_enc1<<<(N_EDGES * 64 + B - 1) / B, B, 0, stream>>>(edge_f, W_e1, b_e1, e_act);

  // edge_mat (bf16) = e @ W_e2 + b_e2 : M=64000, N=4096
  // 4000 mtiles * 64 groups = 256000 waves, 8 waves/block (same mtile per block)
  {
    int groups = 64;                       // 256 Ntiles / NT=4 ; groups % 8 == 0
    int waves  = (N_EDGES / 16) * groups;  // 256000, exactly divisible
    gemm_k64_wmma<4, true, true><<<waves / 8, B, 0, stream>>>(
        e_act, We2sw, b_e2, (void*)emat, N_EDGES, 4096, groups);
  }

  // T rounds of message passing + GRU
  for (int t = 0; t < TSTEPS; ++t) {
    fill_f32<<<(N_NODES * 64 + B - 1) / B, B, 0, stream>>>(mbuf, 0.f, N_NODES * 64);
    msg_kernel<<<N_EDGES / 8, B, 0, stream>>>(emat, hbuf, Esrc, Etgt, mbuf);

    int groups = 3;                         // 12 Ntiles / NT=4 (no A staging: groups%8 != 0)
    int waves  = (N_NODES / 16) * groups;   // 1875
    int blocks = (waves + 7) / 8;
    gemm_k64_wmma<4, false, false><<<blocks, B, 0, stream>>>(
        mbuf, Wihsw, b_ih, (void*)gi, N_NODES, 192, groups);
    gemm_k64_wmma<4, false, false><<<blocks, B, 0, stream>>>(
        hbuf, Whhsw, b_hh, (void*)gh, N_NODES, 192, groups);

    gru_update<<<(N_NODES * 64 + B - 1) / B, B, 0, stream>>>(gi, gh, hbuf);
  }

  // graph readout
  fill_f32<<<1, 64, 0, stream>>>(out, 0.f, N_GRAPHS);
  readout<<<(N_NODES + B - 1) / B, B, 0, stream>>>(hbuf, W_out, b_out, batch, out);
}
